// Attention_82454782148610
// MI455X (gfx1250) — matrix operand order repaired and checked by reading, not compiled, not run on hardware
//
#include <hip/hip_runtime.h>

typedef __attribute__((ext_vector_type(16))) __bf16 v16bf;
typedef __attribute__((ext_vector_type(8)))  float  v8f;
typedef unsigned short ushort_t;

#define N_TOK  2048
#define DMODEL 1024
#define NH     16
#define DH     64
#define QK_SCALE 0.125f   // 1/sqrt(64)

union FragU { v16bf v; uint4 q[2]; };

__device__ __forceinline__ ushort_t f2bf(float f) {
  unsigned u = __builtin_bit_cast(unsigned, f);
  u += 0x7fffu + ((u >> 16) & 1u);      // round-to-nearest-even
  return (ushort_t)(u >> 16);
}

__device__ __forceinline__ v8f wmma_bf16(v16bf a, v16bf b, v8f c) {
  return __builtin_amdgcn_wmma_f32_16x16x32_bf16(false, a, false, b,
                                                 (short)0, c, false, false);
}

// A-matrix 16x32 bf16 fragment (lane = row, interleaved K per ISA layout):
//   elements 0..7  -> k = k0 + 8*half + (0..7)
//   elements 8..15 -> k = k0 + 16 + 8*half + (0..7)
__device__ __forceinline__ v16bf load_a_frag(const ushort_t* base, int row,
                                             int ldk, int k0, int lane) {
  const int half = lane >> 4;
  const ushort_t* p = base + (size_t)row * ldk + k0;
  FragU u;
  u.q[0] = *(const uint4*)(p + 8 * half);
  u.q[1] = *(const uint4*)(p + 16 + 8 * half);
  return u.v;
}

// B-matrix 32x16 bf16 fragment (lane = column n, interleaved K like A fragment):
// base is row-major [n][k] storage of B^T, i.e. B[k][n] = base[n*ldk + k].
__device__ __forceinline__ v16bf load_b_frag(const ushort_t* base, int nrow,
                                             int ldk, int k0, int lane) {
  const int half = lane >> 4;
  const ushort_t* p = base + (size_t)nrow * ldk + k0;
  FragU u;
  u.q[0] = *(const uint4*)(p + 8 * half);
  u.q[1] = *(const uint4*)(p + 16 + 8 * half);
  return u.v;
}

__global__ void cast_f32_bf16(const float* __restrict__ src,
                              ushort_t* __restrict__ dst, int n) {
  int i = blockIdx.x * blockDim.x + threadIdx.x;
  int stride = gridDim.x * blockDim.x;
  for (; i < n; i += stride) dst[i] = f2bf(src[i]);
}

// Shared GEMM mainloop: C[(MT*16)x64] += A[(MT*16)xK] . W[64xK]^T
// Direct global fragment loads (everything is GL2-resident) + prefetch ahead.
template <int MT>
__device__ __forceinline__ void gemm_mainloop(const ushort_t* __restrict__ A,
                                              const ushort_t* __restrict__ W,
                                              int m0, int n0, int kdim,
                                              int lane, v8f c[MT][4]) {
  const int l16 = lane & 15;
  for (int k0 = 0; k0 < kdim; k0 += 32) {
    if (k0 + 128 < kdim) {            // pull lines 4 K-steps ahead into cache
      __builtin_prefetch(A + (size_t)(m0 + lane) * kdim + k0 + 128, 0, 1);
      if (MT == 4)
        __builtin_prefetch(A + (size_t)(m0 + 32 + lane) * kdim + k0 + 128, 0, 1);
      __builtin_prefetch(W + (size_t)(n0 + lane) * kdim + k0 + 128, 0, 1);
      __builtin_prefetch(W + (size_t)(n0 + 32 + lane) * kdim + k0 + 128, 0, 1);
    }
    v16bf a[MT], b[4];
#pragma unroll
    for (int i = 0; i < MT; ++i)
      a[i] = load_a_frag(A, m0 + i * 16 + l16, kdim, k0, lane);
#pragma unroll
    for (int j = 0; j < 4; ++j)
      b[j] = load_b_frag(W, n0 + j * 16 + l16, kdim, k0, lane);
#pragma unroll
    for (int i = 0; i < MT; ++i)
#pragma unroll
      for (int j = 0; j < 4; ++j)
        c[i][j] = wmma_bf16(a[i], b[j], c[i][j]);
  }
}

// ---------------------------------------------------------------------------
// QKV GEMM: [2048x1024]bf16 * [3072x1024]^T bf16 -> bf16 q/k row-major, v^T.
// Block = 4 waves (2x2), each wave a 64x64 tile. Grid = (3072/128, 2048/128).
// ---------------------------------------------------------------------------
__global__ void __launch_bounds__(128)
qkv_gemm(const ushort_t* __restrict__ xb, const ushort_t* __restrict__ wb,
         const float* __restrict__ bias, ushort_t* __restrict__ qws,
         ushort_t* __restrict__ kws, ushort_t* __restrict__ vT) {
  const int lane = threadIdx.x & 31;
  const int wv   = threadIdx.x >> 5;
  const int l16  = lane & 15;
  const int half = lane >> 4;
  const int m0 = blockIdx.y * 128 + (wv >> 1) * 64;
  const int n0 = blockIdx.x * 128 + (wv & 1) * 64;

  v8f c[4][4];
#pragma unroll
  for (int i = 0; i < 4; ++i)
#pragma unroll
    for (int j = 0; j < 4; ++j) c[i][j] = v8f{};

  gemm_mainloop<4>(xb, wb, m0, n0, DMODEL, lane, c);

#pragma unroll
  for (int i = 0; i < 4; ++i) {
#pragma unroll
    for (int j = 0; j < 4; ++j) {
      const int col = n0 + j * 16 + l16;          // 0..3071
      const float bv = bias[col];
#pragma unroll
      for (int r = 0; r < 8; ++r) {
        const int row = m0 + i * 16 + r + 8 * half;
        const ushort_t val = f2bf(c[i][j][r] + bv);
        if (col < DMODEL) {
          qws[(size_t)row * DMODEL + col] = val;
        } else if (col < 2 * DMODEL) {
          kws[(size_t)row * DMODEL + (col - DMODEL)] = val;
        } else {
          // v stored transposed: vT[d_global][token], d_global = h*64+dh
          vT[(size_t)(col - 2 * DMODEL) * N_TOK + row] = val;
        }
      }
    }
  }
}

// ---------------------------------------------------------------------------
// Flash attention: one wave per (head, 32-query block). Online softmax fp32.
// Block = 4 waves. Grid = 16*64/4 = 256.
// ---------------------------------------------------------------------------
__global__ void __launch_bounds__(128)
attn_kernel(const ushort_t* __restrict__ qws, const ushort_t* __restrict__ kws,
            const ushort_t* __restrict__ vT, ushort_t* __restrict__ attws) {
  __shared__ __align__(16) ushort_t pbuf[4][32 * 32];
  const int lane = threadIdx.x & 31;
  const int wv   = threadIdx.x >> 5;
  const int l16  = lane & 15;
  const int half = lane >> 4;
  const int gw = blockIdx.x * 4 + wv;
  const int h  = gw >> 6;
  const int q0 = (gw & 63) << 5;      // 32 queries per wave
  ushort_t* pw = pbuf[wv];

  v16bf aq[2][2];                      // [query-16-block][k-chunk of Dh]
#pragma unroll
  for (int qi = 0; qi < 2; ++qi) {
    aq[qi][0] = load_a_frag(qws, q0 + qi * 16 + l16, DMODEL, h * DH + 0,  lane);
    aq[qi][1] = load_a_frag(qws, q0 + qi * 16 + l16, DMODEL, h * DH + 32, lane);
  }

  float m8[2][8], l8[2][8];
  v8f o[2][4];
#pragma unroll
  for (int qi = 0; qi < 2; ++qi) {
#pragma unroll
    for (int t = 0; t < 4; ++t) o[qi][t] = v8f{};
#pragma unroll
    for (int r = 0; r < 8; ++r) { m8[qi][r] = -3.0e38f; l8[qi][r] = 0.0f; }
  }

  for (int kc = 0; kc < N_TOK; kc += 32) {
    // prefetch next chunk's K rows / V rows one iteration ahead
    if (kc + 32 < N_TOK) {
      __builtin_prefetch(kws + (size_t)(kc + 32 + lane) * DMODEL + h * DH, 0, 1);
      __builtin_prefetch(vT + (size_t)(h * DH + lane) * N_TOK + kc + 32, 0, 1);
      __builtin_prefetch(vT + (size_t)(h * DH + 32 + lane) * N_TOK + kc + 32, 0, 1);
    }

    // K^T B-fragments for keys [kc, kc+32), shared by both query blocks
    const v16bf bk0a = load_b_frag(kws, kc + l16,      DMODEL, h * DH + 0,  lane);
    const v16bf bk0b = load_b_frag(kws, kc + l16,      DMODEL, h * DH + 32, lane);
    const v16bf bk1a = load_b_frag(kws, kc + 16 + l16, DMODEL, h * DH + 0,  lane);
    const v16bf bk1b = load_b_frag(kws, kc + 16 + l16, DMODEL, h * DH + 32, lane);

    v8f s[2][2];
#pragma unroll
    for (int qi = 0; qi < 2; ++qi) {
      s[qi][0] = v8f{};
      s[qi][0] = wmma_bf16(aq[qi][0], bk0a, s[qi][0]);
      s[qi][0] = wmma_bf16(aq[qi][1], bk0b, s[qi][0]);
      s[qi][1] = v8f{};
      s[qi][1] = wmma_bf16(aq[qi][0], bk1a, s[qi][1]);
      s[qi][1] = wmma_bf16(aq[qi][1], bk1b, s[qi][1]);
    }

#pragma unroll
    for (int qi = 0; qi < 2; ++qi) {
#pragma unroll
      for (int r = 0; r < 8; ++r) {
        const float x0 = s[qi][0][r] * QK_SCALE;
        const float x1 = s[qi][1][r] * QK_SCALE;
        float vmax = fmaxf(x0, x1);
#pragma unroll
        for (int msk = 8; msk >= 1; msk >>= 1)
          vmax = fmaxf(vmax, __shfl_xor(vmax, msk, 16));
        const float mn   = fmaxf(m8[qi][r], vmax);
        const float corr = __expf(m8[qi][r] - mn);
        const float p0 = __expf(x0 - mn);
        const float p1 = __expf(x1 - mn);
        float rs = p0 + p1;
#pragma unroll
        for (int msk = 8; msk >= 1; msk >>= 1)
          rs += __shfl_xor(rs, msk, 16);
        m8[qi][r] = mn;
        l8[qi][r] = l8[qi][r] * corr + rs;
#pragma unroll
        for (int t = 0; t < 4; ++t) o[qi][t][r] *= corr;
        const int row = qi * 16 + r + 8 * half;   // C-layout row this lane holds
        pw[row * 32 + l16]      = f2bf(p0);
        pw[row * 32 + 16 + l16] = f2bf(p1);
      }
    }

    // Re-shape P (C-layout) -> A-fragments via per-wave LDS (DS in-order/wave)
    v16bf pa[2];
#pragma unroll
    for (int qi = 0; qi < 2; ++qi) {
      FragU up;
      up.q[0] = *(const uint4*)(pw + (qi * 16 + l16) * 32 + 8 * half);
      up.q[1] = *(const uint4*)(pw + (qi * 16 + l16) * 32 + 16 + 8 * half);
      pa[qi] = up.v;
    }

    // V B-fragments (contiguous in token index), shared by both query blocks
    v16bf bv[4];
#pragma unroll
    for (int t = 0; t < 4; ++t)
      bv[t] = load_b_frag(vT, h * DH + t * 16 + l16, N_TOK, kc, lane);

#pragma unroll
    for (int qi = 0; qi < 2; ++qi)
#pragma unroll
      for (int t = 0; t < 4; ++t)
        o[qi][t] = wmma_bf16(pa[qi], bv[t], o[qi][t]);
  }

#pragma unroll
  for (int qi = 0; qi < 2; ++qi) {
#pragma unroll
    for (int r = 0; r < 8; ++r) {
      const float inv = 1.0f / l8[qi][r];
      const size_t row =
          (size_t)(q0 + qi * 16 + r + 8 * half) * DMODEL + h * DH + l16;
#pragma unroll
      for (int t = 0; t < 4; ++t)
        attws[row + t * 16] = f2bf(o[qi][t][r] * inv);
    }
  }
}

// ---------------------------------------------------------------------------
// Output projection: [2048x1024]bf16 * [1024x1024]^T bf16 + bias -> fp32 out.
// Smaller GEMM: 32x64 wave tile / 64x128 block tile for higher occupancy.
// Grid = (1024/128, 2048/64) = 256 blocks.
// ---------------------------------------------------------------------------
__global__ void __launch_bounds__(128)
proj_gemm(const ushort_t* __restrict__ ab, const ushort_t* __restrict__ wb,
          const float* __restrict__ bias, float* __restrict__ out) {
  const int lane = threadIdx.x & 31;
  const int wv   = threadIdx.x >> 5;
  const int l16  = lane & 15;
  const int half = lane >> 4;
  const int m0 = blockIdx.y * 64 + (wv >> 1) * 32;
  const int n0 = blockIdx.x * 128 + (wv & 1) * 64;

  v8f c[2][4];
#pragma unroll
  for (int i = 0; i < 2; ++i)
#pragma unroll
    for (int j = 0; j < 4; ++j) c[i][j] = v8f{};

  gemm_mainloop<2>(ab, wb, m0, n0, DMODEL, lane, c);

#pragma unroll
  for (int i = 0; i < 2; ++i) {
#pragma unroll
    for (int j = 0; j < 4; ++j) {
      const int col = n0 + j * 16 + l16;
      const float bv = bias[col];
#pragma unroll
      for (int r = 0; r < 8; ++r) {
        const int row = m0 + i * 16 + r + 8 * half;
        out[(size_t)row * DMODEL + col] = c[i][j][r] + bv;
      }
    }
  }
}

extern "C" void kernel_launch(void* const* d_in, const int* in_sizes, int n_in,
                              void* d_out, int out_size, void* d_ws, size_t ws_size,
                              hipStream_t stream) {
  const float* x      = (const float*)d_in[0];
  const float* w_qkv  = (const float*)d_in[1];
  const float* b_qkv  = (const float*)d_in[2];
  const float* w_proj = (const float*)d_in[3];
  const float* b_proj = (const float*)d_in[4];
  float* out = (float*)d_out;

  char* ws = (char*)d_ws;
  size_t off = 0;
  ushort_t* xb    = (ushort_t*)(ws + off); off += (size_t)N_TOK * DMODEL * 2;      // 4 MiB
  ushort_t* wqkvb = (ushort_t*)(ws + off); off += (size_t)3 * DMODEL * DMODEL * 2; // 6 MiB
  ushort_t* wpb   = (ushort_t*)(ws + off); off += (size_t)DMODEL * DMODEL * 2;     // 2 MiB
  ushort_t* qws   = (ushort_t*)(ws + off); off += (size_t)N_TOK * DMODEL * 2;      // 4 MiB
  ushort_t* kws   = (ushort_t*)(ws + off); off += (size_t)N_TOK * DMODEL * 2;      // 4 MiB
  ushort_t* vT    = (ushort_t*)(ws + off); off += (size_t)N_TOK * DMODEL * 2;      // 4 MiB
  ushort_t* attws = xb;  // xb is dead after qkv_gemm; reuse for attention output

  cast_f32_bf16<<<512, 256, 0, stream>>>(x,      xb,    N_TOK * DMODEL);
  cast_f32_bf16<<<512, 256, 0, stream>>>(w_qkv,  wqkvb, 3 * DMODEL * DMODEL);
  cast_f32_bf16<<<512, 256, 0, stream>>>(w_proj, wpb,   DMODEL * DMODEL);

  dim3 gq(3 * DMODEL / 128, N_TOK / 128);
  qkv_gemm<<<gq, 128, 0, stream>>>(xb, wqkvb, b_qkv, qws, kws, vT);

  attn_kernel<<<(NH * (N_TOK / 32)) / 4, 128, 0, stream>>>(qws, kws, vT, attws);

  dim3 gp(DMODEL / 128, N_TOK / 64);
  proj_gemm<<<gp, 128, 0, stream>>>(attws, wpb, b_proj, out);
}
